// CRNN_43224550867774
// MI455X (gfx1250) — matrix operand order repaired and checked by reading, not compiled, not run on hardware
//
#include <hip/hip_runtime.h>
#include <hip/hip_bf16.h>

// CRNN on MI455X (gfx1250): persistent single-WGP recurrence with f16 WMMA,
// f32 accumulate, LDS-resident state, pre-swizzled B-fragment weights in ws.

#define T_STEPS 2048
#define BATCH   128
#define IN_DIM  64
#define H_DIM   128
#define NH      256     // 2*H
#define RS      264     // padded LDS row stride (f16) for hidden state
#define XRS     136     // padded LDS row stride (f16) for x staging

typedef __attribute__((ext_vector_type(16))) _Float16     v16h;
typedef __attribute__((ext_vector_type(8)))  float        v8f;
typedef __attribute__((ext_vector_type(8)))  unsigned int v8u;

// ---- workspace layout (dword offsets). Each fragment block is
// (kchunk, ctile, lane, j): 32 lanes x 8 dwords = 1KB, lane-contiguous. ----
#define FRAG_IH0 0          // 4  kchunks (K=128)  -> 16384 dwords
#define FRAG_HH0 16384      // 8  kchunks (K=256)  -> 32768
#define FRAG_L1  49152      // 16 kchunks (K=512: ih1 then hh1) -> 65536
#define FRAG_OUT 114688     // 8  kchunks (K=256)  -> 32768
#define BIAS_OFF 147456     // 3*256 f32 combined biases
// total ws need: (147456 + 768)*4 bytes ~= 579 KB

// expanded complex weight element: M[k][n], w = [2][H][Kin] (Wr, Wi)
__device__ __forceinline__ float cw(const float* w, int Kin, int K, int N) {
  int kr = K % Kin, kp = K / Kin;
  int nr = N & (H_DIM - 1), np = N >> 7;
  float wr = w[nr * Kin + kr];
  float wi = w[H_DIM * Kin + nr * Kin + kr];
  return (kp == np) ? wr : ((kp == 1) ? -wi : wi);
}

__device__ __forceinline__ v8f wmma_f16(v8u a, v8u b, v8f c) {
  return __builtin_amdgcn_wmma_f32_16x16x32_f16(
      false, __builtin_bit_cast(v16h, a),
      false, __builtin_bit_cast(v16h, b),
      (short)0, c, false, false);
}

// B fragment: pre-swizzled in ws, 8 contiguous dwords per lane (2x b128)
__device__ __forceinline__ v8u load_bfrag(const unsigned int* frag, int kc, int c, int lane) {
  return *(const v8u*)(frag + (((kc << 4) + c) << 8) + (lane << 3));
}

// A fragment from row-major f16 LDS: lane holds row M=lane&15, packed K pairs
__device__ __forceinline__ v8u load_afrag(const _Float16* base, int rowStride,
                                          int row0, int kbase, int lane) {
  int m = lane & 15, half = lane >> 4;
  const _Float16* rp = base + (row0 + m) * rowStride;
  v8u a;
#pragma unroll
  for (int j = 0; j < 8; ++j) {
    int k = kbase + ((j & 4) << 2) + ((j & 3) << 1) + (half << 3);
    a[j] = *(const unsigned int*)(rp + k);
  }
  return a;
}

__device__ __forceinline__ void vzero4(v8f acc[4]) {
#pragma unroll
  for (int cc = 0; cc < 4; ++cc)
#pragma unroll
    for (int j = 0; j < 8; ++j) acc[cc][j] = 0.0f;
}

// ---------------- weight expansion + swizzle into B-fragment order ----------
__global__ void crnn_prep_weights(unsigned int* __restrict__ ws,
                                  const float* __restrict__ wih0,
                                  const float* __restrict__ whh0,
                                  const float* __restrict__ wih1,
                                  const float* __restrict__ whh1,
                                  const float* __restrict__ wout) {
  int idx = blockIdx.x * blockDim.x + threadIdx.x;
  if (idx >= BIAS_OFF) return;
  const float* w; int Kin, local, kc, kcSub;
  if (idx < FRAG_HH0)      { local = idx;            kc = local >> 12; w = wih0; Kin = IN_DIM; kcSub = kc; }
  else if (idx < FRAG_L1)  { local = idx - FRAG_HH0; kc = local >> 12; w = whh0; Kin = H_DIM;  kcSub = kc; }
  else if (idx < FRAG_OUT) { local = idx - FRAG_L1;  kc = local >> 12; Kin = H_DIM;
                             if (kc < 8) { w = wih1; kcSub = kc; } else { w = whh1; kcSub = kc - 8; } }
  else                     { local = idx - FRAG_OUT; kc = local >> 12; w = wout; Kin = H_DIM;  kcSub = kc; }
  int rem  = local & 4095;
  int c    = rem >> 8;
  int r2   = rem & 255;
  int lane = r2 >> 3;
  int j    = r2 & 7;
  int K = 32 * kcSub + ((j & 4) << 2) + ((j & 3) << 1) + 8 * (lane >> 4);   // B layout K pair
  int N = 16 * c + (lane & 15);
  _Float16 lo = (_Float16)cw(w, Kin, K, N);
  _Float16 hi = (_Float16)cw(w, Kin, K + 1, N);
  _Float16* p = (_Float16*)&ws[idx];
  p[0] = lo; p[1] = hi;
}

__global__ void crnn_prep_bias(unsigned int* __restrict__ ws,
                               const float* __restrict__ bih0, const float* __restrict__ bhh0,
                               const float* __restrict__ bih1, const float* __restrict__ bhh1,
                               const float* __restrict__ bout) {
  int i = blockIdx.x * blockDim.x + threadIdx.x;  // 0..767
  if (i >= 768) return;
  int which = i >> 8, n = i & 255;
  int np = n >> 7, nr = n & 127;
  float v;
  if (which == 0)      v = bih0[np * 128 + nr] + bhh0[np * 128 + nr];
  else if (which == 1) v = bih1[np * 128 + nr] + bhh1[np * 128 + nr];
  else                 v = bout[np * 128 + nr];
  ((float*)(ws + BIAS_OFF))[i] = v;
}

// ---------------- parallel prefold: u[t] = x[t] @ M_ih0 (no recurrence) -----
__global__ __launch_bounds__(1024) void crnn_prefold(const float* __restrict__ x,
                                                     const unsigned int* __restrict__ ws,
                                                     float* __restrict__ uy) {
  __shared__ _Float16 xs[BATCH][XRS];
  int t = blockIdx.x;
  const float* xt = x + (size_t)t * BATCH * 128;
  for (int i = threadIdx.x; i < BATCH * 128; i += 1024)
    xs[i >> 7][i & 127] = (_Float16)xt[i];
  __syncthreads();

  int tid = threadIdx.x, lane = tid & 31, wave = tid >> 5;
  int row0 = (wave >> 2) << 4, c0 = (wave & 3) << 2;
  int m = lane & 15, half = lane >> 4;

  v8f acc[4];
  vzero4(acc);
#pragma unroll
  for (int kc = 0; kc < 4; ++kc) {
    v8u a = load_afrag(&xs[0][0], XRS, row0, kc << 5, lane);
#pragma unroll
    for (int cc = 0; cc < 4; ++cc) {
      v8u b = load_bfrag(ws + FRAG_IH0, kc, c0 + cc, lane);
      acc[cc] = wmma_f16(a, b, acc[cc]);
    }
  }
  float* ut = uy + (size_t)t * BATCH * NH;
#pragma unroll
  for (int cc = 0; cc < 4; ++cc)
#pragma unroll
    for (int j = 0; j < 8; ++j)
      ut[(row0 + j + (half << 3)) * NH + ((c0 + cc) << 4) + m] = acc[cc][j];
}

// ---------------- persistent sequential recurrence: 1 workgroup, 32 waves ---
__global__ __launch_bounds__(1024) void crnn_steps(float* __restrict__ out,
                                                   const unsigned int* __restrict__ ws,
                                                   const float* __restrict__ h0) {
  __shared__ _Float16 hbuf[2][BATCH][RS];   // f16 hidden state, padded rows
  int tid = threadIdx.x, lane = tid & 31, wave = tid >> 5;
  int row0 = (wave >> 2) << 4, c0 = (wave & 3) << 2;
  int m = lane & 15, half = lane >> 4;

  for (int i = tid; i < 2 * BATCH * NH; i += 1024) {
    int layer = i >> 15, rem = i & 32767;
    hbuf[layer][rem >> 8][rem & 255] = (_Float16)h0[i];
  }
  const float* bias = (const float*)(ws + BIAS_OFF);
  float b0r[4], b1r[4], bor[4];
#pragma unroll
  for (int cc = 0; cc < 4; ++cc) {
    int col = ((c0 + cc) << 4) + m;
    b0r[cc] = bias[col]; b1r[cc] = bias[256 + col]; bor[cc] = bias[512 + col];
  }
  __syncthreads();

  for (int t = 0; t < T_STEPS; ++t) {
    float* yt = out + (size_t)t * BATCH * NH;

    // prefetch x-part u[t] (staged in y region by crnn_prefold)
    float ureg[4][8];
#pragma unroll
    for (int cc = 0; cc < 4; ++cc)
#pragma unroll
      for (int j = 0; j < 8; ++j)
        ureg[cc][j] = yt[(row0 + j + (half << 3)) * NH + ((c0 + cc) << 4) + m];

    // ---- layer 0: acc = h0_state @ M_hh0 ----
    v8f acc[4];
    vzero4(acc);
#pragma unroll
    for (int kc = 0; kc < 8; ++kc) {
      v8u a = load_afrag(&hbuf[0][0][0], RS, row0, kc << 5, lane);
#pragma unroll
      for (int cc = 0; cc < 4; ++cc) {
        v8u b = load_bfrag(ws + FRAG_HH0, kc, c0 + cc, lane);
        acc[cc] = wmma_f16(a, b, acc[cc]);
      }
    }
    __syncthreads();  // all reads of hbuf[0] complete
#pragma unroll
    for (int cc = 0; cc < 4; ++cc)
#pragma unroll
      for (int j = 0; j < 8; ++j) {
        int row = row0 + j + (half << 3);
        int col = ((c0 + cc) << 4) + m;
        hbuf[0][row][col] = (_Float16)tanhf(acc[cc][j] + ureg[cc][j] + b0r[cc]);
      }
    __syncthreads();  // new h0 visible

    // ---- layer 1 (fused K=512): [new_h0 | h1_state] @ [M_ih1 ; M_hh1] ----
    vzero4(acc);
#pragma unroll
    for (int kc = 0; kc < 16; ++kc) {
      const _Float16* src = (kc < 8) ? &hbuf[0][0][0] : &hbuf[1][0][0];
      v8u a = load_afrag(src, RS, row0, (kc & 7) << 5, lane);
#pragma unroll
      for (int cc = 0; cc < 4; ++cc) {
        v8u b = load_bfrag(ws + FRAG_L1, kc, c0 + cc, lane);
        acc[cc] = wmma_f16(a, b, acc[cc]);
      }
    }
    __syncthreads();  // all reads of hbuf[0]/hbuf[1] complete
#pragma unroll
    for (int cc = 0; cc < 4; ++cc)
#pragma unroll
      for (int j = 0; j < 8; ++j) {
        int row = row0 + j + (half << 3);
        int col = ((c0 + cc) << 4) + m;
        hbuf[1][row][col] = (_Float16)tanhf(acc[cc][j] + b1r[cc]);
      }
    __syncthreads();  // new h1 visible

    // ---- output projection: y[t] = tanh(h1 @ M_out + b_out) ----
    vzero4(acc);
#pragma unroll
    for (int kc = 0; kc < 8; ++kc) {
      v8u a = load_afrag(&hbuf[1][0][0], RS, row0, kc << 5, lane);
#pragma unroll
      for (int cc = 0; cc < 4; ++cc) {
        v8u b = load_bfrag(ws + FRAG_OUT, kc, c0 + cc, lane);
        acc[cc] = wmma_f16(a, b, acc[cc]);
      }
    }
#pragma unroll
    for (int cc = 0; cc < 4; ++cc)
#pragma unroll
      for (int j = 0; j < 8; ++j) {
        int row = row0 + j + (half << 3);
        int col = ((c0 + cc) << 4) + m;
        yt[row * NH + col] = tanhf(acc[cc][j] + bor[cc]);
      }
    // next iteration's hbuf writes are barrier-protected; no extra sync needed
  }
  __syncthreads();

  // final hidden state hn -> d_out tail
  float* hn = out + (size_t)T_STEPS * BATCH * NH;
  for (int i = tid; i < 2 * BATCH * NH; i += 1024) {
    int layer = i >> 15, rem = i & 32767;
    hn[i] = (float)hbuf[layer][rem >> 8][rem & 255];
  }
}

extern "C" void kernel_launch(void* const* d_in, const int* in_sizes, int n_in,
                              void* d_out, int out_size, void* d_ws, size_t ws_size,
                              hipStream_t stream) {
  (void)in_sizes; (void)n_in; (void)out_size; (void)ws_size;
  const float* x      = (const float*)d_in[0];
  const float* h0     = (const float*)d_in[1];
  const float* wih_w0 = (const float*)d_in[2];
  const float* wih_b0 = (const float*)d_in[3];
  const float* whh_w0 = (const float*)d_in[4];
  const float* whh_b0 = (const float*)d_in[5];
  const float* wih_w1 = (const float*)d_in[6];
  const float* wih_b1 = (const float*)d_in[7];
  const float* whh_w1 = (const float*)d_in[8];
  const float* whh_b1 = (const float*)d_in[9];
  const float* out_w  = (const float*)d_in[10];
  const float* out_b  = (const float*)d_in[11];
  unsigned int* ws = (unsigned int*)d_ws;
  float* out = (float*)d_out;

  crnn_prep_weights<<<576, 256, 0, stream>>>(ws, wih_w0, whh_w0, wih_w1, whh_w1, out_w);
  crnn_prep_bias<<<3, 256, 0, stream>>>(ws, wih_b0, whh_b0, wih_b1, whh_b1, out_b);
  crnn_prefold<<<T_STEPS, 1024, 0, stream>>>(x, ws, out);   // u[t] staged into y region
  crnn_steps<<<1, 1024, 0, stream>>>(out, ws, h0);          // persistent recurrence
}
